// MultiHeadAttention_88536455840315
// MI455X (gfx1250) — matrix operand order repaired and checked by reading, not compiled, not run on hardware
//
#include <hip/hip_runtime.h>

// ---------------- problem constants ----------------
#define B_    2
#define N_    2048
#define E_    768
#define H_    12
#define D_    64
#define QKVC  2304            // 3 * H_ * D_
#define SCALE 0.125f          // 64^-0.5
#define LOG2PI 1.8378770664093453f

typedef __attribute__((ext_vector_type(16))) __bf16 v16bf;
typedef __attribute__((ext_vector_type(8)))  float  v8f;
typedef unsigned int  v4u  __attribute__((ext_vector_type(4)));
typedef int           v8i_t __attribute__((ext_vector_type(8)));
typedef int           v4i_t __attribute__((ext_vector_type(4)));
typedef unsigned short ushort_t;
typedef unsigned long long u64_t;

union BF16Tile { uint4 q[2]; v16bf v; };
union BF16x8   { ushort_t u[8]; uint4 q; };

__device__ __forceinline__ ushort_t f2bf(float f) {
  unsigned u = __float_as_uint(f);
  unsigned r = (u + 0x7FFFu + ((u >> 16) & 1u)) >> 16;   // RNE
  return (ushort_t)r;
}

__device__ __forceinline__ v8f vzero8() {
  v8f z = {0.f,0.f,0.f,0.f,0.f,0.f,0.f,0.f};
  return z;
}

__device__ __forceinline__ v8f wmma_bf16(v16bf a, v16bf b, v8f c) {
  // D = A(16x32 bf16) * B(32x16 bf16) + C(16x16 f32)
  return __builtin_amdgcn_wmma_f32_16x16x32_bf16(
      /*neg_a=*/false, a, /*neg_b=*/false, b,
      /*c_mod=*/(short)0, c, /*reuse_a=*/false, /*reuse_b=*/false);
}

// A-tile (16 rows x 32 K) from row-major [rows x ld] bf16.
// lane<16: row=row0+lr, K = k0+[0..7], k0+[16..23]; lane>=16: K = k0+[8..15], k0+[24..31]
__device__ __forceinline__ v16bf load_a_tile(const ushort_t* base, int ld,
                                             int row0, int k0, int lr, int hl) {
  const ushort_t* p = base + (size_t)(row0 + lr) * ld + k0 + hl * 8;
  BF16Tile t;
  t.q[0] = *(const uint4*)p;
  t.q[1] = *(const uint4*)(p + 16);
  return t.v;
}

// B-tile (32 K x 16 cols), storage "per-column contiguous K":
// element (k, n) at base[(col0+n)*ld + k]. lane n<16: K=k0+[0..7], k0+[16..23]; lane n+16: K=k0+[8..15], k0+[24..31]
__device__ __forceinline__ v16bf load_b_tile(const ushort_t* base, int ld,
                                             int col0, int k0, int lr, int hl) {
  const ushort_t* p = base + (size_t)(col0 + lr) * ld + k0 + hl * 8;
  BF16Tile t;
  t.q[0] = *(const uint4*)p;
  t.q[1] = *(const uint4*)(p + 16);
  return t.v;
}

__device__ __forceinline__ float redmax16(float x) {
  x = fmaxf(x, __shfl_xor(x, 1, 32));
  x = fmaxf(x, __shfl_xor(x, 2, 32));
  x = fmaxf(x, __shfl_xor(x, 4, 32));
  x = fmaxf(x, __shfl_xor(x, 8, 32));
  return x;
}
__device__ __forceinline__ float redsum16(float x) {
  x += __shfl_xor(x, 1, 32);
  x += __shfl_xor(x, 2, 32);
  x += __shfl_xor(x, 4, 32);
  x += __shfl_xor(x, 8, 32);
  return x;
}

// ---------------- Tensor Data Mover: DMA a 2-D bf16 tile (rows x width) into LDS ----------------
// D# built per CDNA5 ISA:
//  group0: [1:0]=count(1), [63:32]=lds_addr, [120:64]=global_addr, [127:126]=type(2)
//  group1: [17:16]=data_size(1 => 2B), [79:48]=tensor_dim0, [111:80]=tensor_dim1,
//          [127:112]=tile_dim0, [143:128]=tile_dim1, [207:160]=tensor_dim0_stride
#define HAVE_TDM (__has_builtin(__builtin_amdgcn_tensor_load_to_lds))

__device__ __forceinline__ void tdm_load_2d_bf16(const ushort_t* gsrc, unsigned lds_byte_off,
                                                 unsigned width_elems, unsigned rows,
                                                 unsigned row_stride_elems) {
#if HAVE_TDM
  union { u64_t q[2]; v4u v; } g0;
  g0.q[0] = 1ull | ((u64_t)lds_byte_off << 32);
  g0.q[1] = (((u64_t)(uintptr_t)gsrc) & ((1ull << 57) - 1)) | (2ull << 62);

  union { u64_t q[4]; v8i_t v; } g1;
  u64_t td0 = width_elems, td1 = rows;
  u64_t t0  = width_elems, t1  = rows;
  u64_t s0  = row_stride_elems;
  g1.q[0] = 0x00010000ull | ((td0 & 0xffffull) << 48);          // data_size=1 (2B); td0[15:0]
  g1.q[1] = (td0 >> 16) | (td1 << 16) | (t0 << 48);             // td0[31:16], td1, tile_dim0
  g1.q[2] = t1 | ((s0 & 0xffffffffull) << 32);                  // tile_dim1, stride0[31:0]
  g1.q[3] = (s0 >> 32);                                          // stride0[47:32]; stride1=0
  v4i_t gz4 = {0, 0, 0, 0};
  v8i_t gz8 = {0, 0, 0, 0, 0, 0, 0, 0};
  // clang-23 / therock-10.0 lane: 6-arg form (g0, g1, g2, g3, g4, cpol)
  __builtin_amdgcn_tensor_load_to_lds(g0.v, g1.v, gz4, gz4, gz8, 0);
#else
  (void)gsrc; (void)lds_byte_off; (void)width_elems; (void)rows; (void)row_stride_elems;
#endif
}

__device__ __forceinline__ void tdm_wait0() {
#if __has_builtin(__builtin_amdgcn_s_wait_tensorcnt)
  __builtin_amdgcn_s_wait_tensorcnt(0);
#else
  asm volatile("s_wait_tensorcnt 0x0" ::: "memory");
#endif
}

// Stage a 16 x E_ bf16 A-strip into dynamic LDS (As). wave0 issues the DMA.
__device__ __forceinline__ void stage_a_strip(ushort_t* As, const ushort_t* gsrc) {
#if HAVE_TDM
  if (threadIdx.x < 32) {
    tdm_load_2d_bf16(gsrc, 0u, E_, 16u, E_);
    tdm_wait0();
  }
  asm volatile("" ::: "memory");
  __syncthreads();
#else
  // cooperative fallback: 16*E_ bf16 = 1536 uint4, 128 threads x 12
  const uint4* s = (const uint4*)gsrc;
  uint4* d = (uint4*)As;
#pragma unroll
  for (int i = 0; i < 12; ++i) d[threadIdx.x + 128 * i] = s[threadIdx.x + 128 * i];
  __syncthreads();
#endif
}

// ---------------- kernel 0a: fp32 -> bf16 elementwise ----------------
__global__ void __launch_bounds__(256)
cvt_bf16_kernel(const float* __restrict__ src, ushort_t* __restrict__ dst, int count) {
  int i = blockIdx.x * 256 + threadIdx.x;
  if (i < count) dst[i] = f2bf(src[i]);
}

// ---------------- kernel 0b: fp32 (rows x cols) -> bf16 transposed ----------------
__global__ void __launch_bounds__(256)
cvt_bf16_T_kernel(const float* __restrict__ src, ushort_t* __restrict__ dst,
                  int rows, int cols) {
  int i = blockIdx.x * 256 + threadIdx.x;
  if (i < rows * cols) {
    int r = i / cols, c = i - r * cols;
    dst[(size_t)c * rows + r] = f2bf(src[i]);
  }
}

// ---------------- kernel 1: QKV projection (x @ w_qkv) -> q/k/vT bf16 ----------------
// A strip (shared by all 4 waves of the block) staged via TDM into LDS.
__global__ void __launch_bounds__(128)
qkv_gemm_kernel(const ushort_t* __restrict__ xb, const ushort_t* __restrict__ wT,
                ushort_t* __restrict__ qd, ushort_t* __restrict__ kd,
                ushort_t* __restrict__ vtd) {
  extern __shared__ __align__(16) ushort_t As[];   // 16 x E_ bf16 = 24KB
  const int lane = threadIdx.x & 31, w = threadIdx.x >> 5;
  const int lr = lane & 15, hl = lane >> 4;
  const int TN = QKVC / 64;                        // 36 (divisible by 4 -> same row0 per block)
  int tilebase = blockIdx.x * 4;
  int tm = tilebase / TN, tn = tilebase - tm * TN + w;
  int row0 = tm * 16, col0 = tn * 64;

  stage_a_strip(As, xb + (size_t)row0 * E_);

  v8f acc[4];
#pragma unroll
  for (int t = 0; t < 4; ++t) acc[t] = vzero8();

  for (int kk = 0; kk < E_; kk += 32) {
    // A fragment from LDS (ds_load_b128 x2)
    const ushort_t* ap = As + lr * E_ + kk + hl * 8;
    BF16Tile at;
    at.q[0] = *(const uint4*)ap;
    at.q[1] = *(const uint4*)(ap + 16);
#pragma unroll
    for (int t = 0; t < 4; ++t)
      acc[t] = wmma_bf16(at.v, load_b_tile(wT, E_, col0 + 16 * t, kk, lr, hl), acc[t]);
  }

  // epilogue: the 64-wide tile is exactly one (section, head)
  int sec = col0 / E_;
  int cc  = col0 - sec * E_;
  int h   = cc / D_;
  int b   = row0 / N_;
  int n0  = row0 - b * N_;

#pragma unroll
  for (int t = 0; t < 4; ++t) {
    int d = 16 * t + lr;
    if (sec == 0) {
      ushort_t* dst = qd + (((size_t)b * H_ + h) * N_ + n0 + 8 * hl) * D_ + d;
#pragma unroll
      for (int g = 0; g < 8; ++g) dst[(size_t)g * D_] = f2bf(acc[t][g] * SCALE);
    } else if (sec == 1) {
      ushort_t* dst = kd + (((size_t)b * H_ + h) * N_ + n0 + 8 * hl) * D_ + d;
#pragma unroll
      for (int g = 0; g < 8; ++g) dst[(size_t)g * D_] = f2bf(acc[t][g]);
    } else {
      BF16x8 tmp;
#pragma unroll
      for (int g = 0; g < 8; ++g) tmp.u[g] = f2bf(acc[t][g]);
      *(uint4*)(vtd + (((size_t)b * H_ + h) * D_ + d) * N_ + n0 + 8 * hl) = tmp.q;
    }
  }
}

// ---------------- kernel 2: flash attention, product = softmax(q k^T) @ v ----------------
__global__ void __launch_bounds__(128)
attn_kernel(const ushort_t* __restrict__ q, const ushort_t* __restrict__ k,
            const ushort_t* __restrict__ vt, float* __restrict__ prod) {
  __shared__ __align__(16) ushort_t Ps[4][16 * 32];   // per-wave P transpose scratch

  const int lane = threadIdx.x & 31, w = threadIdx.x >> 5;
  const int lr = lane & 15, hl = lane >> 4;
  const int nblk = N_ / 64;
  int bh = blockIdx.x / nblk;
  int i0 = (blockIdx.x - bh * nblk) * 64 + w * 16;

  const ushort_t* qb = q  + (size_t)bh * N_ * D_;
  const ushort_t* kb = k  + (size_t)bh * N_ * D_;
  const ushort_t* vb = vt + (size_t)bh * D_ * N_;

  v16bf qa0 = load_a_tile(qb, D_, i0, 0,  lr, hl);
  v16bf qa1 = load_a_tile(qb, D_, i0, 32, lr, hl);

  v8f o0 = vzero8(), o1 = vzero8(), o2 = vzero8(), o3 = vzero8();
  const v8f vz = vzero8();
  float m_r[8], l_r[8];
#pragma unroll
  for (int g = 0; g < 8; ++g) { m_r[g] = -1e30f; l_r[g] = 0.f; }

  ushort_t* ps = &Ps[w][0];

  for (int jc = 0; jc < N_; jc += 32) {
    if (jc + 32 < N_) {
      __builtin_prefetch(kb + (size_t)(jc + 32 + lr) * D_, 0, 3);
      __builtin_prefetch(vb + (size_t)lr * N_ + jc + 32, 0, 3);
    }
    // scores: S = q @ k^T (two 16x16 tiles, K=64 split into 2 wmmas each)
    v8f s0 = wmma_bf16(qa0, load_b_tile(kb, D_, jc,      0,  lr, hl), vz);
    s0     = wmma_bf16(qa1, load_b_tile(kb, D_, jc,      32, lr, hl), s0);
    v8f s1 = wmma_bf16(qa0, load_b_tile(kb, D_, jc + 16, 0,  lr, hl), vz);
    s1     = wmma_bf16(qa1, load_b_tile(kb, D_, jc + 16, 32, lr, hl), s1);

    // online softmax (row M = g + 8*hl, col = lr / 16+lr)
    float corr[8];
#pragma unroll
    for (int g = 0; g < 8; ++g) {
      float mx = redmax16(fmaxf(s0[g], s1[g]));
      float mnew = fmaxf(m_r[g], mx);
      corr[g] = __expf(m_r[g] - mnew);
      m_r[g] = mnew;
      float p0 = __expf(s0[g] - mnew);
      float p1 = __expf(s1[g] - mnew);
      s0[g] = p0; s1[g] = p1;
      l_r[g] = l_r[g] * corr[g] + redsum16(p0 + p1);
    }
#pragma unroll
    for (int g = 0; g < 8; ++g) {
      o0[g] *= corr[g]; o1[g] *= corr[g]; o2[g] *= corr[g]; o3[g] *= corr[g];
    }

    // P: C-layout -> LDS (16x32 bf16) -> A-layout (wave-private; LDS in-order per wave)
#pragma unroll
    for (int g = 0; g < 8; ++g) {
      int r = g + 8 * hl;
      ps[r * 32 + lr]      = f2bf(s0[g]);
      ps[r * 32 + 16 + lr] = f2bf(s1[g]);
    }
    v16bf pa;
    {
      const ushort_t* pp = ps + lr * 32 + hl * 8;
      BF16Tile t;
      t.q[0] = *(const uint4*)pp;
      t.q[1] = *(const uint4*)(pp + 16);
      pa = t.v;
    }

    // product accumulation: o += P(16x32) @ V(32x64)
    o0 = wmma_bf16(pa, load_b_tile(vb, N_, 0,  jc, lr, hl), o0);
    o1 = wmma_bf16(pa, load_b_tile(vb, N_, 16, jc, lr, hl), o1);
    o2 = wmma_bf16(pa, load_b_tile(vb, N_, 32, jc, lr, hl), o2);
    o3 = wmma_bf16(pa, load_b_tile(vb, N_, 48, jc, lr, hl), o3);
  }

  float* pr = prod + ((size_t)bh * N_ + i0) * D_;
#pragma unroll
  for (int g = 0; g < 8; ++g) {
    float inv = 1.0f / l_r[g];
    int r = g + 8 * hl;
    pr[(size_t)r * D_ + lr]      = o0[g] * inv;
    pr[(size_t)r * D_ + 16 + lr] = o1[g] * inv;
    pr[(size_t)r * D_ + 32 + lr] = o2[g] * inv;
    pr[(size_t)r * D_ + 48 + lr] = o3[g] * inv;
  }
}

// ---------------- kernel 3: cross-head stats + log-prob weighting ----------------
__global__ void __launch_bounds__(64)
stats_weight_kernel(const float* __restrict__ prod, ushort_t* __restrict__ wtd) {
  int bn = blockIdx.x;
  int b = bn / N_, n = bn - b * N_;
  int d = threadIdx.x, w = d >> 5;

  float p[H_];
  float s = 0.f;
#pragma unroll
  for (int h = 0; h < H_; ++h) {
    p[h] = prod[(((size_t)b * H_ + h) * N_ + n) * D_ + d];
    s += p[h];
  }
  float mean = s * (1.0f / H_);
  float ss = 0.f;
#pragma unroll
  for (int h = 0; h < H_; ++h) { float dv = p[h] - mean; ss += dv * dv; }
  float var = ss * (1.0f / (H_ - 1));          // unbiased, ddof=1
  float base = -0.5f * LOG2PI - __logf(var);   // per-d term of the reference's algebra
  float inv4v = 0.25f / var;

  __shared__ float part[H_][2];
#pragma unroll
  for (int h = 0; h < H_; ++h) {
    float dv = p[h] - mean;
    float t = base + dv * dv * inv4v;
    t += __shfl_xor(t, 1, 32);
    t += __shfl_xor(t, 2, 32);
    t += __shfl_xor(t, 4, 32);
    t += __shfl_xor(t, 8, 32);
    t += __shfl_xor(t, 16, 32);
    if ((threadIdx.x & 31) == 0) part[h][w] = t;
  }
  __syncthreads();

  ushort_t* wrow = wtd + ((size_t)b * N_ + n) * E_;
#pragma unroll
  for (int h = 0; h < H_; ++h) {
    float lp = part[h][0] + part[h][1];        // fixed order -> deterministic
    wrow[h * D_ + d] = f2bf(lp * p[h]);
  }
}

// ---------------- kernel 4: out = weighted @ w_out + b_out ----------------
__global__ void __launch_bounds__(128)
out_gemm_kernel(const ushort_t* __restrict__ wtd, const ushort_t* __restrict__ woT,
                const float* __restrict__ bias, float* __restrict__ out) {
  extern __shared__ __align__(16) ushort_t As[];   // 16 x E_ bf16
  const int lane = threadIdx.x & 31, w = threadIdx.x >> 5;
  const int lr = lane & 15, hl = lane >> 4;
  const int TN = E_ / 64;                          // 12 (divisible by 4)
  int tilebase = blockIdx.x * 4;
  int tm = tilebase / TN, tn = tilebase - tm * TN + w;
  int row0 = tm * 16, col0 = tn * 64;

  stage_a_strip(As, wtd + (size_t)row0 * E_);

  v8f acc[4];
#pragma unroll
  for (int t = 0; t < 4; ++t) acc[t] = vzero8();

  for (int kk = 0; kk < E_; kk += 32) {
    const ushort_t* ap = As + lr * E_ + kk + hl * 8;
    BF16Tile at;
    at.q[0] = *(const uint4*)ap;
    at.q[1] = *(const uint4*)(ap + 16);
#pragma unroll
    for (int t = 0; t < 4; ++t)
      acc[t] = wmma_bf16(at.v, load_b_tile(woT, E_, col0 + 16 * t, kk, lr, hl), acc[t]);
  }

#pragma unroll
  for (int t = 0; t < 4; ++t) {
    int col = col0 + 16 * t + lr;
    float bv = bias[col];
    float* dst = out + (size_t)(row0 + 8 * hl) * E_ + col;
#pragma unroll
    for (int g = 0; g < 8; ++g) dst[(size_t)g * E_] = acc[t][g] + bv;
  }
}

// ---------------- host launcher ----------------
extern "C" void kernel_launch(void* const* d_in, const int* in_sizes, int n_in,
                              void* d_out, int out_size, void* d_ws, size_t ws_size,
                              hipStream_t stream) {
  (void)in_sizes; (void)n_in; (void)out_size; (void)ws_size;
  const float* x     = (const float*)d_in[0];   // (B,N,E)
  const float* w_qkv = (const float*)d_in[1];   // (E, QKVC)
  const float* w_out = (const float*)d_in[2];   // (E, E)
  const float* b_out = (const float*)d_in[3];   // (E)
  float* out = (float*)d_out;

  char* ws = (char*)d_ws;
  auto align256 = [](size_t v) { return (v + 255) & ~(size_t)255; };
  size_t off = 0;
  ushort_t* xb    = (ushort_t*)(ws + off); off += align256((size_t)B_ * N_ * E_ * 2);
  ushort_t* wqkvT = (ushort_t*)(ws + off); off += align256((size_t)E_ * QKVC * 2);
  ushort_t* woutT = (ushort_t*)(ws + off); off += align256((size_t)E_ * E_ * 2);
  ushort_t* qd    = (ushort_t*)(ws + off); off += align256((size_t)B_ * H_ * N_ * D_ * 2);
  ushort_t* kd    = (ushort_t*)(ws + off); off += align256((size_t)B_ * H_ * N_ * D_ * 2);
  ushort_t* vtd   = (ushort_t*)(ws + off); off += align256((size_t)B_ * H_ * N_ * D_ * 2);
  float*    prod  = (float*)   (ws + off); off += align256((size_t)B_ * H_ * N_ * D_ * 4);
  ushort_t* wtd   = (ushort_t*)(ws + off); off += align256((size_t)B_ * N_ * E_ * 2);

  // 0) precision conversion (weights transposed for contiguous B-tile loads)
  {
    int cnt = B_ * N_ * E_;
    cvt_bf16_kernel<<<(cnt + 255) / 256, 256, 0, stream>>>(x, xb, cnt);
  }
  {
    int cnt = E_ * QKVC;
    cvt_bf16_T_kernel<<<(cnt + 255) / 256, 256, 0, stream>>>(w_qkv, wqkvT, E_, QKVC);
  }
  {
    int cnt = E_ * E_;
    cvt_bf16_T_kernel<<<(cnt + 255) / 256, 256, 0, stream>>>(w_out, woutT, E_, E_);
  }

  const size_t a_lds = (size_t)16 * E_ * 2;   // 24KB dynamic LDS A-strip

  // 1) QKV projection: (4096/16)*(2304/64) = 9216 wave-tiles, 4 waves/block
  qkv_gemm_kernel<<<9216 / 4, 128, a_lds, stream>>>(xb, wqkvT, qd, kd, vtd);

  // 2) attention: B*H*(N/64) blocks of 4 waves
  attn_kernel<<<B_ * H_ * (N_ / 64), 128, 0, stream>>>(qd, kd, vtd, prod);

  // 3) cross-head stats + weighting: one block per (b,n)
  stats_weight_kernel<<<B_ * N_, 64, 0, stream>>>(prod, wtd);

  // 4) output projection: (4096/16)*(768/64) = 3072 wave-tiles
  out_gemm_kernel<<<3072 / 4, 128, a_lds, stream>>>(wtd, woutT, b_out, out);
}
